// DualBranchFusionModel_14439680049641
// MI455X (gfx1250) — matrix operand order repaired and checked by reading, not compiled, not run on hardware
//
#include <hip/hip_runtime.h>
#include <hip/hip_bf16.h>

typedef __attribute__((ext_vector_type(16))) _Float16 v16h;
typedef __attribute__((ext_vector_type(8)))  float    v8f;

#define HID 128

// ---------------------------------------------------------------------------
// WMMA GEMM: Out[M,Nout] = A[M,K] @ W[K,Nout] (+ bias [+ BN + ReLU])
// A: f32 row-major. Wt: f16 column-major ([Nout][K]).
// One wave computes a 16x64 output block (4 x 16x16 WMMA accumulators),
// reusing the A-fragment across the 4 column tiles.
// Requires: K % 32 == 0, M % 16 == 0, Nout % 64 == 0.
// ---------------------------------------------------------------------------
__global__ void wmma_gemm_kernel(const float* __restrict__ A,
                                 const _Float16* __restrict__ Wt,
                                 float* __restrict__ Out,
                                 int M, int K, int Nout,
                                 const float* __restrict__ bias,
                                 const float* __restrict__ bng,
                                 const float* __restrict__ bnb,
                                 const float* __restrict__ bnm,
                                 const float* __restrict__ bnv,
                                 int fuse_bn_relu)
{
    const int lane        = threadIdx.x & 31;
    const int waveInBlk   = threadIdx.x >> 5;
    const int wavesPerBlk = blockDim.x >> 5;
    const long waveId     = (long)blockIdx.x * wavesPerBlk + waveInBlk;
    const int colGroups   = Nout >> 6;                  // 64-wide column groups
    const long numTiles   = (long)(M >> 4) * colGroups;
    if (waveId >= numTiles) return;             // wave-uniform exit: EXEC stays full

    const int rowTile = (int)(waveId / colGroups);
    const int colGrp  = (int)(waveId % colGroups);
    const int laneHi  = lane >> 4;
    const int laneLo  = lane & 15;
    const int rowBase = rowTile << 4;
    const int colBase = (colGrp << 6) + laneLo;   // lane's column in tile c is colBase+16c
    const int arow    = rowBase + laneLo;         // this lane's A row

    v8f acc0 = {0.f,0.f,0.f,0.f,0.f,0.f,0.f,0.f};
    v8f acc1 = acc0, acc2 = acc0, acc3 = acc0;

    const int ksteps = K >> 5;
    const float* Arow = A + (long)arow * K;
    const _Float16* Wc0 = Wt + (long)(colBase +  0) * K;
    const _Float16* Wc1 = Wt + (long)(colBase + 16) * K;
    const _Float16* Wc2 = Wt + (long)(colBase + 32) * K;
    const _Float16* Wc3 = Wt + (long)(colBase + 48) * K;

    for (int kk = 0; kk < ksteps; ++kk) {
        const int kbase = kk << 5;
        // A fragment: lanes 0-15 hold K {0..15},
        // lanes 16-31 hold K {16..31}; convert f32 -> f16 in regs.
        const v8f a0 = *(const v8f*)(Arow + kbase + laneHi * 16);
        const v8f a1 = *(const v8f*)(Arow + kbase + laneHi * 16 + 8);
        v16h af;
        #pragma unroll
        for (int i = 0; i < 8; ++i) {
            af[i]     = (_Float16)a0[i];
            af[8 + i] = (_Float16)a1[i];
        }
        // B fragments: each lane holds 16 contiguous K-halves of its column.
        const int kb = kbase + laneHi * 16;
        const v16h bf0 = *(const v16h*)(Wc0 + kb);
        const v16h bf1 = *(const v16h*)(Wc1 + kb);
        const v16h bf2 = *(const v16h*)(Wc2 + kb);
        const v16h bf3 = *(const v16h*)(Wc3 + kb);
        acc0 = __builtin_amdgcn_wmma_f32_16x16x32_f16(false, af, false, bf0, (short)0, acc0, false, false);
        acc1 = __builtin_amdgcn_wmma_f32_16x16x32_f16(false, af, false, bf1, (short)0, acc1, false, false);
        acc2 = __builtin_amdgcn_wmma_f32_16x16x32_f16(false, af, false, bf2, (short)0, acc2, false, false);
        acc3 = __builtin_amdgcn_wmma_f32_16x16x32_f16(false, af, false, bf3, (short)0, acc3, false, false);
    }

    v8f accs[4] = {acc0, acc1, acc2, acc3};
    if (fuse_bn_relu) {
        #pragma unroll
        for (int c = 0; c < 4; ++c) {
            const int col   = colBase + c * 16;
            const float b     = bias[col];
            const float scale = bng[col] * rsqrtf(bnv[col] + 1e-5f);
            const float shift = bnb[col] + (b - bnm[col]) * scale;
            #pragma unroll
            for (int r = 0; r < 8; ++r) {
                const int row = rowBase + r + laneHi * 8;
                float v = accs[c][r] * scale + shift;
                Out[(long)row * Nout + col] = v > 0.f ? v : 0.f;
            }
        }
    } else {
        #pragma unroll
        for (int c = 0; c < 4; ++c) {
            const int col = colBase + c * 16;
            const float b = bias[col];
            #pragma unroll
            for (int r = 0; r < 8; ++r) {
                const int row = rowBase + r + laneHi * 8;
                Out[(long)row * Nout + col] = accs[c][r] + b;
            }
        }
    }
}

// ---------------------------------------------------------------------------
// Convert a batch of row-major f32 weights [K,Nout] to f16 column-major [Nout,K]
// ---------------------------------------------------------------------------
__global__ void convert_weights_kernel(const float* __restrict__ W,
                                       _Float16* __restrict__ Wt,
                                       int K, int Nout, long total)
{
    long i = (long)blockIdx.x * blockDim.x + threadIdx.x;
    if (i >= total) return;
    const long per = (long)K * Nout;
    const long s = i / per;
    const long r = i % per;
    const int n = (int)(r / K);
    const int k = (int)(r % K);
    Wt[s * per + (long)n * K + k] = (_Float16)W[s * per + (long)k * Nout + n];
}

__global__ void zero_kernel(float* __restrict__ p, long n)
{
    long i = (long)blockIdx.x * blockDim.x + threadIdx.x;
    if (i < n) p[i] = 0.f;
}

// h = x @ aW + ab : [N,9] @ [9,128]
__global__ void atom_encoder_kernel(const float* __restrict__ X,
                                    const float* __restrict__ W,
                                    const float* __restrict__ b,
                                    float* __restrict__ H, int N)
{
    long i = (long)blockIdx.x * blockDim.x + threadIdx.x;
    if (i >= (long)N * HID) return;
    const int n = (int)(i >> 7);
    const int j = (int)(i & (HID - 1));
    float acc = b[j];
    #pragma unroll
    for (int k = 0; k < 9; ++k) acc += X[n * 9 + k] * W[k * HID + j];
    H[i] = acc;
}

// zin = (1+eps)*h
__global__ void combine_kernel(const float* __restrict__ h,
                               float* __restrict__ zin,
                               const float* __restrict__ eps, long n)
{
    long i = (long)blockIdx.x * blockDim.x + threadIdx.x;
    if (i >= n) return;
    zin[i] = (1.0f + eps[0]) * h[i];
}

// wave per edge: zin[dst] += h[src]  (128 features, 4 per lane)
__global__ void edge_scatter_kernel(const float* __restrict__ H,
                                    const int* __restrict__ ei,
                                    float* __restrict__ Z, int E)
{
    const int lane = threadIdx.x & 31;
    const int wave = blockIdx.x * (blockDim.x >> 5) + (threadIdx.x >> 5);
    if (wave >= E) return;
    const int s = ei[wave];
    const int d = ei[E + wave];
    const float4 hv = *(const float4*)(H + (long)s * HID + lane * 4);
    float* zp = Z + (long)d * HID + lane * 4;
    atomicAdd(zp + 0, hv.x);
    atomicAdd(zp + 1, hv.y);
    atomicAdd(zp + 2, hv.z);
    atomicAdd(zp + 3, hv.w);
}

// wave per node: sums[bat[n]] += h[n]; cnt[bat[n]] += 1
__global__ void pool_scatter_kernel(const float* __restrict__ H,
                                    const int* __restrict__ bat,
                                    float* __restrict__ sums,
                                    float* __restrict__ cnt, int N)
{
    const int lane = threadIdx.x & 31;
    const int wave = blockIdx.x * (blockDim.x >> 5) + (threadIdx.x >> 5);
    if (wave >= N) return;
    const int b = bat[wave];
    const float4 hv = *(const float4*)(H + (long)wave * HID + lane * 4);
    float* sp = sums + (long)b * HID + lane * 4;
    atomicAdd(sp + 0, hv.x);
    atomicAdd(sp + 1, hv.y);
    atomicAdd(sp + 2, hv.z);
    atomicAdd(sp + 3, hv.w);
    if (lane == 0) atomicAdd(&cnt[b], 1.0f);
}

__global__ void pool_div_kernel(float* __restrict__ emb,
                                const float* __restrict__ cnt, int B)
{
    long i = (long)blockIdx.x * blockDim.x + threadIdx.x;
    if (i >= (long)B * HID) return;
    const int b = (int)(i >> 7);
    emb[i] /= fmaxf(cnt[b], 1.0f);
}

// In-place LayerNorm + ReLU, one wave per row
__global__ void ln_relu_kernel(float* __restrict__ X,
                               const float* __restrict__ g,
                               const float* __restrict__ b,
                               int rows, int ncols)
{
    const int lane = threadIdx.x & 31;
    const int wave = blockIdx.x * (blockDim.x >> 5) + (threadIdx.x >> 5);
    if (wave >= rows) return;
    float* row = X + (long)wave * ncols;
    float s = 0.f, s2 = 0.f;
    for (int c = lane; c < ncols; c += 32) {
        const float x = row[c];
        s += x; s2 += x * x;
    }
    #pragma unroll
    for (int off = 16; off > 0; off >>= 1) {
        s  += __shfl_xor(s,  off, 32);
        s2 += __shfl_xor(s2, off, 32);
    }
    const float inv = 1.0f / (float)ncols;
    const float mu  = s * inv;
    const float var = s2 * inv - mu * mu;
    const float rs  = rsqrtf(var + 1e-5f);
    for (int c = lane; c < ncols; c += 32) {
        float x = (row[c] - mu) * rs * g[c] + b[c];
        row[c] = x > 0.f ? x : 0.f;
    }
}

// dst[r, doff:doff+scols] = src[r, :scols]
__global__ void copy_cols_kernel(const float* __restrict__ src, int scols,
                                 float* __restrict__ dst, int dcols, int doff,
                                 int rows)
{
    long i = (long)blockIdx.x * blockDim.x + threadIdx.x;
    if (i >= (long)rows * scols) return;
    const int r = (int)(i / scols);
    const int c = (int)(i % scols);
    dst[(long)r * dcols + doff + c] = src[i];
}

// temp = relu(t @ W1 + b1) @ W2 + b2 : [B,1] -> [B,32] -> [B,32]
__global__ void temp_mlp_kernel(const float* __restrict__ t,
                                const float* __restrict__ W1,
                                const float* __restrict__ b1,
                                const float* __restrict__ W2,
                                const float* __restrict__ b2,
                                float* __restrict__ out, int B)
{
    long i = (long)blockIdx.x * blockDim.x + threadIdx.x;
    if (i >= (long)B * 32) return;
    const int bi = (int)(i >> 5);
    const int j  = (int)(i & 31);
    const float tv = t[bi];
    float acc = b2[j];
    #pragma unroll
    for (int k = 0; k < 32; ++k) {
        float hk = tv * W1[k] + b1[k];
        hk = hk > 0.f ? hk : 0.f;
        acc += hk * W2[k * 32 + j];
    }
    out[i] = acc;
}

// out[b] = h2[b,:] @ W3 + b3 : [B,64] -> [B,1]
__global__ void final_dot_kernel(const float* __restrict__ h2,
                                 const float* __restrict__ W3,
                                 const float* __restrict__ b3,
                                 float* __restrict__ out, int B)
{
    int bi = blockIdx.x * blockDim.x + threadIdx.x;
    if (bi >= B) return;
    float acc = b3[0];
    #pragma unroll
    for (int k = 0; k < 64; ++k) acc += h2[(long)bi * 64 + k] * W3[k];
    out[bi] = acc;
}

// ---------------------------------------------------------------------------
static inline int divUp(long a, long b) { return (int)((a + b - 1) / b); }

extern "C" void kernel_launch(void* const* d_in, const int* in_sizes, int n_in,
                              void* d_out, int out_size, void* d_ws, size_t ws_size,
                              hipStream_t stream)
{
    // ---- inputs (setup_inputs order) ----
    const float* dx = (const float*)d_in[0];
    const int*   de = (const int*)  d_in[1];
    const int*   db = (const int*)  d_in[2];
    const float* sx = (const float*)d_in[3];
    const int*   se = (const int*)  d_in[4];
    const int*   sb = (const int*)  d_in[5];
    const float* t  = (const float*)d_in[6];
    const float* enc_atom_W = (const float*)d_in[7];   // [3,9,128]
    const float* enc_atom_b = (const float*)d_in[8];   // [3,128]
    const float* enc_eps    = (const float*)d_in[9];   // [3,4]
    const float* enc_W1     = (const float*)d_in[10];  // [3,4,128,256]
    const float* enc_b1     = (const float*)d_in[11];  // [3,4,256]
    const float* enc_bn1_g  = (const float*)d_in[12];
    const float* enc_bn1_b  = (const float*)d_in[13];
    const float* enc_bn1_m  = (const float*)d_in[14];
    const float* enc_bn1_v  = (const float*)d_in[15];
    const float* enc_W2     = (const float*)d_in[16];  // [3,4,256,128]
    const float* enc_b2     = (const float*)d_in[17];  // [3,4,128]
    const float* enc_bn_g   = (const float*)d_in[18];
    const float* enc_bn_b   = (const float*)d_in[19];
    const float* enc_bn_m   = (const float*)d_in[20];
    const float* enc_bn_v   = (const float*)d_in[21];
    const float* temp_W1    = (const float*)d_in[22];
    const float* temp_b1    = (const float*)d_in[23];
    const float* temp_W2    = (const float*)d_in[24];
    const float* temp_b2    = (const float*)d_in[25];
    const float* int_W      = (const float*)d_in[26];  // [256,128]
    const float* int_b      = (const float*)d_in[27];
    const float* int_ln_g   = (const float*)d_in[28];
    const float* int_ln_b   = (const float*)d_in[29];
    const float* fus_W1     = (const float*)d_in[30];  // [288,128]
    const float* fus_b1     = (const float*)d_in[31];
    const float* fus_ln1_g  = (const float*)d_in[32];
    const float* fus_ln1_b  = (const float*)d_in[33];
    const float* fus_W2     = (const float*)d_in[34];  // [128,64]
    const float* fus_b2     = (const float*)d_in[35];
    const float* fus_ln2_g  = (const float*)d_in[36];
    const float* fus_ln2_b  = (const float*)d_in[37];
    const float* fus_W3     = (const float*)d_in[38];  // [64,1]
    const float* fus_b3     = (const float*)d_in[39];

    const int N = in_sizes[0] / 9;   // 50000 (multiple of 16)
    const int E = in_sizes[1] / 2;   // 600000
    const int B = in_sizes[6];       // 4096

    // ---- workspace carve-up (256B aligned slabs) ----
    char* wp = (char*)d_ws;
    auto carve = [&](size_t bytes) -> void* {
        void* p = (void*)wp;
        wp += (bytes + 255) & ~(size_t)255;
        return p;
    };
    float* h      = (float*)carve((size_t)N * HID * 4);
    float* zin    = (float*)carve((size_t)N * HID * 4);
    float* z1     = (float*)carve((size_t)N * 2 * HID * 4);
    float* emb    = (float*)carve((size_t)3 * B * HID * 4);
    float* cnt    = (float*)carve((size_t)B * 4);
    _Float16* w1h   = (_Float16*)carve((size_t)12 * HID * 2 * HID * 2);
    _Float16* w2h   = (_Float16*)carve((size_t)12 * 2 * HID * HID * 2);
    _Float16* intWh = (_Float16*)carve((size_t)2 * HID * HID * 2);
    _Float16* fW1h  = (_Float16*)carve((size_t)288 * HID * 2);
    _Float16* fW2h  = (_Float16*)carve((size_t)HID * 64 * 2);
    float* bs_so  = (float*)carve((size_t)B * 2 * HID * 4);
    float* inter  = (float*)carve((size_t)B * HID * 4);
    float* temp32 = (float*)carve((size_t)B * 32 * 4);
    float* fusion = (float*)carve((size_t)B * 288 * 4);
    float* h1     = (float*)carve((size_t)B * HID * 4);
    float* h2     = (float*)carve((size_t)B * 64 * 4);

    const int TPB = 256;

    // ---- one-time (per launch) weight conversion to f16 col-major ----
    {
        long tot = (long)12 * HID * 2 * HID;
        convert_weights_kernel<<<divUp(tot, TPB), TPB, 0, stream>>>(
            enc_W1, w1h, HID, 2 * HID, tot);
        convert_weights_kernel<<<divUp(tot, TPB), TPB, 0, stream>>>(
            enc_W2, w2h, 2 * HID, HID, tot);
        long t2 = (long)2 * HID * HID;
        convert_weights_kernel<<<divUp(t2, TPB), TPB, 0, stream>>>(
            int_W, intWh, 2 * HID, HID, t2);
        long t3 = (long)288 * HID;
        convert_weights_kernel<<<divUp(t3, TPB), TPB, 0, stream>>>(
            fus_W1, fW1h, 288, HID, t3);
        long t4 = (long)HID * 64;
        convert_weights_kernel<<<divUp(t4, TPB), TPB, 0, stream>>>(
            fus_W2, fW2h, HID, 64, t4);
    }

    // ---- 3 GIN encoders ----
    const long nh = (long)N * HID;
    for (int e = 0; e < 3; ++e) {
        const float* X   = (e == 2) ? sx : dx;
        const int*   EI  = (e == 2) ? se : de;
        const int*   BAT = (e == 2) ? sb : db;

        atom_encoder_kernel<<<divUp(nh, TPB), TPB, 0, stream>>>(
            X, enc_atom_W + (long)e * 9 * HID, enc_atom_b + (long)e * HID, h, N);

        for (int l = 0; l < 4; ++l) {
            const long sl = (long)e * 4 + l;
            // zin = (1+eps)*h ; zin += scatter(h[src] -> dst)
            combine_kernel<<<divUp(nh, TPB), TPB, 0, stream>>>(
                h, zin, enc_eps + sl, nh);
            edge_scatter_kernel<<<divUp((long)E, TPB / 32), TPB, 0, stream>>>(
                h, EI, zin, E);
            // z1 = relu(BN1(zin @ W1 + b1))  [N,128]->[N,256]
            {
                long tiles = (long)(N / 16) * (2 * HID / 64);
                wmma_gemm_kernel<<<divUp(tiles, TPB / 32), TPB, 0, stream>>>(
                    zin, w1h + sl * HID * 2 * HID, z1, N, HID, 2 * HID,
                    enc_b1 + sl * 2 * HID, enc_bn1_g + sl * 2 * HID,
                    enc_bn1_b + sl * 2 * HID, enc_bn1_m + sl * 2 * HID,
                    enc_bn1_v + sl * 2 * HID, 1);
            }
            // h = relu(BN2(z1 @ W2 + b2))  [N,256]->[N,128]
            {
                long tiles = (long)(N / 16) * (HID / 64);
                wmma_gemm_kernel<<<divUp(tiles, TPB / 32), TPB, 0, stream>>>(
                    z1, w2h + sl * 2 * HID * HID, h, N, 2 * HID, HID,
                    enc_b2 + sl * HID, enc_bn_g + sl * HID,
                    enc_bn_b + sl * HID, enc_bn_m + sl * HID,
                    enc_bn_v + sl * HID, 1);
            }
        }

        // global mean pool -> emb[e]
        float* embE = emb + (long)e * B * HID;
        zero_kernel<<<divUp((long)B * HID, TPB), TPB, 0, stream>>>(embE, (long)B * HID);
        zero_kernel<<<divUp((long)B, TPB), TPB, 0, stream>>>(cnt, B);
        pool_scatter_kernel<<<divUp((long)N, TPB / 32), TPB, 0, stream>>>(
            h, BAT, embE, cnt, N);
        pool_div_kernel<<<divUp((long)B * HID, TPB), TPB, 0, stream>>>(embE, cnt, B);
    }

    float* aq  = emb;                       // encoder 0
    float* bs_ = emb + (long)B * HID;       // encoder 1
    float* so  = emb + (long)2 * B * HID;   // encoder 2

    // inter = relu(LN(concat(bs, so) @ int_W + int_b))
    copy_cols_kernel<<<divUp((long)B * HID, TPB), TPB, 0, stream>>>(
        bs_, HID, bs_so, 2 * HID, 0, B);
    copy_cols_kernel<<<divUp((long)B * HID, TPB), TPB, 0, stream>>>(
        so, HID, bs_so, 2 * HID, HID, B);
    {
        long tiles = (long)(B / 16) * (HID / 64);
        wmma_gemm_kernel<<<divUp(tiles, TPB / 32), TPB, 0, stream>>>(
            bs_so, intWh, inter, B, 2 * HID, HID,
            int_b, nullptr, nullptr, nullptr, nullptr, 0);
    }
    ln_relu_kernel<<<divUp((long)B, TPB / 32), TPB, 0, stream>>>(
        inter, int_ln_g, int_ln_b, B, HID);

    // temp MLP [B,1] -> [B,32]
    temp_mlp_kernel<<<divUp((long)B * 32, TPB), TPB, 0, stream>>>(
        t, temp_W1, temp_b1, temp_W2, temp_b2, temp32, B);

    // fusion = concat([aq, inter, temp])  [B,288]
    copy_cols_kernel<<<divUp((long)B * HID, TPB), TPB, 0, stream>>>(
        aq, HID, fusion, 288, 0, B);
    copy_cols_kernel<<<divUp((long)B * HID, TPB), TPB, 0, stream>>>(
        inter, HID, fusion, 288, HID, B);
    copy_cols_kernel<<<divUp((long)B * 32, TPB), TPB, 0, stream>>>(
        temp32, 32, fusion, 288, 2 * HID, B);

    // h1 = relu(LN(fusion @ fus_W1 + fus_b1))  [B,288]->[B,128]
    {
        long tiles = (long)(B / 16) * (HID / 64);
        wmma_gemm_kernel<<<divUp(tiles, TPB / 32), TPB, 0, stream>>>(
            fusion, fW1h, h1, B, 288, HID,
            fus_b1, nullptr, nullptr, nullptr, nullptr, 0);
    }
    ln_relu_kernel<<<divUp((long)B, TPB / 32), TPB, 0, stream>>>(
        h1, fus_ln1_g, fus_ln1_b, B, HID);

    // h2 = relu(LN(h1 @ fus_W2 + fus_b2))  [B,128]->[B,64]
    {
        long tiles = (long)(B / 16) * (64 / 64);
        wmma_gemm_kernel<<<divUp(tiles, TPB / 32), TPB, 0, stream>>>(
            h1, fW2h, h2, B, HID, 64,
            fus_b2, nullptr, nullptr, nullptr, nullptr, 0);
    }
    ln_relu_kernel<<<divUp((long)B, TPB / 32), TPB, 0, stream>>>(
        h2, fus_ln2_g, fus_ln2_b, B, 64);

    // out = h2 @ fus_W3 + fus_b3  [B,1]
    final_dot_kernel<<<divUp((long)B, TPB), TPB, 0, stream>>>(
        h2, fus_W3, fus_b3, (float*)d_out, B);
}
